// OptimizedMoELayer_43508018708619
// MI455X (gfx1250) — compile-verified, matrix operand order checked by reading, not run on hardware
//
#include <hip/hip_runtime.h>

// ---------------- CDNA5 WMMA types ----------------
typedef __attribute__((ext_vector_type(16))) __bf16 v16bf;
typedef __attribute__((ext_vector_type(8)))  float  v8f;

// ---------------- problem constants ----------------
#define TOK   8192
#define DM    1024
#define NE    8
#define HH    2730
#define HP    2752      // H padded to multiple of 64 (and of 32 for the K loop)
#define CAP   1024      // capacity per expert (T / E * 1.0)
#define TK    16384     // T * top_k
#define LDSS  40        // LDS row stride in bf16 elems (80B: staggers banks, keeps 16B align)

// Packed fp32->bf16 (RNE) on the VALU: one instruction for two converts.
__device__ __forceinline__ unsigned int cvt_pk_bf16(float a, float b) {
    unsigned int r;
    asm("v_cvt_pk_bf16_f32 %0, %1, %2" : "=v"(r) : "v"(a), "v"(b));
    return r;
}

__device__ __forceinline__ unsigned short f32_to_bf16(float f) {
    return (unsigned short)(cvt_pk_bf16(f, f) & 0xFFFFu);
}

// Pack 4 floats -> 4 bf16 (8 bytes) at dst (8B-aligned).
__device__ __forceinline__ void store_bf16x4(unsigned short* dst, float a, float b, float c, float d) {
    *reinterpret_cast<unsigned int*>(dst)     = cvt_pk_bf16(a, b);
    *reinterpret_cast<unsigned int*>(dst + 2) = cvt_pk_bf16(c, d);
}

__device__ __forceinline__ v8f zero8() {
    v8f z = {0.f, 0.f, 0.f, 0.f, 0.f, 0.f, 0.f, 0.f};
    return z;
}

// One loader for both A (16x32) and B (32x16) bf16 fragments: row-major tile in
// LDS, lane%16 selects the row, lane/16 selects the K-half (0/8); each lane
// reads K-chunks [kb, kb+8) and [kb+16, kb+24) -> two ds_load_b128.
__device__ __forceinline__ v16bf load_frag(const unsigned short* tile, int rowBase, int lane) {
    const int r  = lane & 15;
    const int kb = (lane >> 4) << 3;
    const unsigned short* p = tile + (rowBase + r) * LDSS + kb;
    union { uint4 q[2]; v16bf v; } f;
    f.q[0] = *reinterpret_cast<const uint4*>(p);
    f.q[1] = *reinterpret_cast<const uint4*>(p + 16);
    return f.v;
}

__device__ __forceinline__ v8f wmma_bf16(v16bf a, v16bf b, v8f c) {
    return __builtin_amdgcn_wmma_f32_16x16x32_bf16(false, a, false, b, (short)0, c, false, false);
}

// ---------------- 1. router: logits, top-2, softmax ----------------
__global__ __launch_bounds__(256) void router_topk_kernel(
        const float* __restrict__ x, const float* __restrict__ gate_w,
        int* __restrict__ topk_idx, float* __restrict__ topk_p) {
    const int lane = threadIdx.x & 31;
    const int t = blockIdx.x * 8 + (threadIdx.x >> 5);   // one wave32 per token
    if (t >= TOK) return;
    const float* xr = x + (size_t)t * DM;
    float acc[NE];
#pragma unroll
    for (int e = 0; e < NE; ++e) acc[e] = 0.f;
    for (int d = lane; d < DM; d += 32) {
        float xv = xr[d];
#pragma unroll
        for (int e = 0; e < NE; ++e) acc[e] += xv * gate_w[e * DM + d];
    }
#pragma unroll
    for (int e = 0; e < NE; ++e) {
#pragma unroll
        for (int off = 16; off > 0; off >>= 1) acc[e] += __shfl_xor(acc[e], off, 32);
    }
    if (lane == 0) {
        int i0 = 0; float l0 = acc[0];
#pragma unroll
        for (int e = 1; e < NE; ++e) if (acc[e] > l0) { l0 = acc[e]; i0 = e; }   // ties -> lowest idx
        int i1 = -1; float l1 = -__builtin_inff();
#pragma unroll
        for (int e = 0; e < NE; ++e) if (e != i0 && acc[e] > l1) { l1 = acc[e]; i1 = e; }
        float m  = fmaxf(l0, l1);
        float e0 = __expf(l0 - m), e1 = __expf(l1 - m);
        float s  = e0 + e1;
        topk_idx[t * 2 + 0] = i0;  topk_idx[t * 2 + 1] = i1;
        topk_p[t * 2 + 0] = e0 / s;  topk_p[t * 2 + 1] = e1 / s;
    }
}

// ---------------- 2. capacity keep (exact reference semantics) ----------------
__global__ __launch_bounds__(256) void capacity_keep_kernel(
        const int* __restrict__ idx, const float* __restrict__ p,
        int* __restrict__ keep, float* __restrict__ slot_w) {
    const int i  = blockIdx.x * 256 + threadIdx.x;
    const int ei = idx[i];
    const float pi = p[i];
    __shared__ int   se[256];
    __shared__ float sp[256];
    int cnt = 0;
    for (int base = 0; base < TK; base += 256) {
        __syncthreads();
        se[threadIdx.x] = idx[base + threadIdx.x];
        sp[threadIdx.x] = p[base + threadIdx.x];
        __syncthreads();
#pragma unroll 8
        for (int j = 0; j < 256; ++j) {
            int jj = base + j;
            cnt += (se[j] == ei) & ((sp[j] > pi) | ((sp[j] == pi) & (jj < i)));
        }
    }
    const int k = (cnt < CAP) ? 1 : 0;
    keep[i] = k;
    slot_w[i] = k ? pi : 0.0f;
}

// ---------------- 3a. init per-expert token lists (padding -> token 0, weight 0) ----
__global__ __launch_bounds__(256) void init_lists_kernel(int* __restrict__ tok_list) {
    int i = blockIdx.x * 256 + threadIdx.x;
    if (i < NE * CAP) tok_list[i] = 0;
}

// ---------------- 3b. deterministic positions + gather lists ----------------
__global__ __launch_bounds__(256) void positions_kernel(
        const int* __restrict__ idx, const int* __restrict__ keep,
        int* __restrict__ slot_e, int* __restrict__ slot_pos, int* __restrict__ tok_list) {
    const int i  = blockIdx.x * 256 + threadIdx.x;
    const int ei = idx[i];
    __shared__ int se[256];
    __shared__ int sk[256];
    int cnt = 0;
    const int lastBase = blockIdx.x * 256;
    for (int base = 0; base <= lastBase; base += 256) {
        __syncthreads();
        se[threadIdx.x] = idx[base + threadIdx.x];
        sk[threadIdx.x] = keep[base + threadIdx.x];
        __syncthreads();
#pragma unroll 8
        for (int j = 0; j < 256; ++j) {
            int jj = base + j;
            cnt += (jj < i) & (se[j] == ei) & sk[j];
        }
    }
    if (keep[i]) {
        slot_e[i]   = ei;
        slot_pos[i] = cnt;                    // position in expert list, token order
        tok_list[ei * CAP + cnt] = i >> 1;    // token index
    } else {
        slot_e[i]   = 0;
        slot_pos[i] = 0;                      // weight is 0 -> contributes nothing
    }
}

// ---------------- 4. stage 1: hidden = silu(Xg Wg^T) * (Xg Wu^T), bf16 out ------
// grid: (HP/64, CAP/128, NE), block 256 (8 waves). 128x64x32 tiles, double-buffered
// LDS with register prefetch; each wave owns a 32x32 block of both G and U -> 8 WMMA
// per K-step against 12 ds_load_b128.
__global__ __launch_bounds__(256) void moe_gate_up_kernel(
        const float* __restrict__ x, const float* __restrict__ w_gate,
        const float* __restrict__ w_up, const int* __restrict__ tok_list,
        unsigned short* __restrict__ hidden) {
    const int e  = blockIdx.z;
    const int m0 = blockIdx.y * 128;  // capacity slots
    const int n0 = blockIdx.x * 64;   // hidden units (HP space)
    __shared__ __align__(16) unsigned short Xs[2][128 * LDSS];
    __shared__ __align__(16) unsigned short Gs[2][64 * LDSS];
    __shared__ __align__(16) unsigned short Us[2][64 * LDSS];
    const int tid  = threadIdx.x;
    const int lane = tid & 31;
    const int wid  = tid >> 5;
    const int wm   = wid & 3;         // 4 row-blocks of 32
    const int wn   = wid >> 2;        // 2 col-blocks of 32

    v8f ag[2][2], au[2][2];
#pragma unroll
    for (int mi = 0; mi < 2; ++mi)
#pragma unroll
        for (int ni = 0; ni < 2; ++ni) { ag[mi][ni] = zero8(); au[mi][ni] = zero8(); }

    int trow[4];
#pragma unroll
    for (int l = 0; l < 4; ++l)
        trow[l] = tok_list[e * CAP + m0 + ((tid + l * 256) >> 3)];

    float4 xr[4], gr[2], ur[2];
    auto loadG = [&](int k) {
#pragma unroll
        for (int l = 0; l < 4; ++l) {
            const int c4 = ((tid + l * 256) & 7) * 4;
            xr[l] = *reinterpret_cast<const float4*>(x + (size_t)trow[l] * DM + k + c4);
        }
#pragma unroll
        for (int l = 0; l < 2; ++l) {
            const int idx  = tid + l * 256;
            const int hrow = n0 + (idx >> 3);
            const int c4   = (idx & 7) * 4;
            if (hrow < HH) {
                const size_t wo = ((size_t)e * HH + hrow) * DM + k + c4;
                gr[l] = *reinterpret_cast<const float4*>(w_gate + wo);
                ur[l] = *reinterpret_cast<const float4*>(w_up + wo);
            } else {
                gr[l] = make_float4(0.f, 0.f, 0.f, 0.f);
                ur[l] = make_float4(0.f, 0.f, 0.f, 0.f);
            }
        }
    };
    auto storeL = [&](int buf) {
#pragma unroll
        for (int l = 0; l < 4; ++l) {
            const int idx = tid + l * 256;
            store_bf16x4(&Xs[buf][(idx >> 3) * LDSS + (idx & 7) * 4],
                         xr[l].x, xr[l].y, xr[l].z, xr[l].w);
        }
#pragma unroll
        for (int l = 0; l < 2; ++l) {
            const int idx = tid + l * 256;
            const int off = (idx >> 3) * LDSS + (idx & 7) * 4;
            store_bf16x4(&Gs[buf][off], gr[l].x, gr[l].y, gr[l].z, gr[l].w);
            store_bf16x4(&Us[buf][off], ur[l].x, ur[l].y, ur[l].z, ur[l].w);
        }
    };

    loadG(0);
    storeL(0);
    int buf = 0;
    constexpr int NT = DM / 32;
    for (int kt = 0; kt < NT; ++kt) {
        __syncthreads();                          // stores to lds[buf] visible
        const bool more = (kt + 1 < NT);
        if (more) loadG((kt + 1) * 32);           // global prefetch overlaps WMMA
        const v16bf a0 = load_frag(Xs[buf], wm * 32,      lane);
        const v16bf a1 = load_frag(Xs[buf], wm * 32 + 16, lane);
        const v16bf g0 = load_frag(Gs[buf], wn * 32,      lane);
        const v16bf g1 = load_frag(Gs[buf], wn * 32 + 16, lane);
        const v16bf u0 = load_frag(Us[buf], wn * 32,      lane);
        const v16bf u1 = load_frag(Us[buf], wn * 32 + 16, lane);
        ag[0][0] = wmma_bf16(a0, g0, ag[0][0]);
        ag[0][1] = wmma_bf16(a0, g1, ag[0][1]);
        ag[1][0] = wmma_bf16(a1, g0, ag[1][0]);
        ag[1][1] = wmma_bf16(a1, g1, ag[1][1]);
        au[0][0] = wmma_bf16(a0, u0, au[0][0]);
        au[0][1] = wmma_bf16(a0, u1, au[0][1]);
        au[1][0] = wmma_bf16(a1, u0, au[1][0]);
        au[1][1] = wmma_bf16(a1, u1, au[1][1]);
        if (more) storeL(buf ^ 1);                // other buffer: safe after barrier
        buf ^= 1;
    }

    // epilogue: SiLU(g)*u -> bf16 hidden[e][slot][HP] (HP tail stays zero via W guard)
#pragma unroll
    for (int mi = 0; mi < 2; ++mi) {
        const int rbase = m0 + wm * 32 + mi * 16 + ((lane >> 4) << 3);
#pragma unroll
        for (int ni = 0; ni < 2; ++ni) {
            const int cb = n0 + wn * 32 + ni * 16 + (lane & 15);
#pragma unroll
            for (int i = 0; i < 8; ++i) {
                const float g = ag[mi][ni][i], u = au[mi][ni][i];
                const float h = (g / (1.0f + __expf(-g))) * u;
                hidden[((size_t)e * CAP + rbase + i) * HP + cb] = f32_to_bf16(h);
            }
        }
    }
}

// ---------------- 5. stage 2: down_out = hidden @ Wd^T (per slot, unweighted) ---
// grid: (DM/64, CAP/128, NE), block 256. Same 128x64x32 double-buffered scheme,
// 4 WMMA per K-step per wave.
__global__ __launch_bounds__(256) void moe_down_kernel(
        const unsigned short* __restrict__ hidden, const float* __restrict__ w_down,
        float* __restrict__ down_out) {
    const int e  = blockIdx.z;
    const int m0 = blockIdx.y * 128;  // capacity slots
    const int n0 = blockIdx.x * 64;   // output d dims
    __shared__ __align__(16) unsigned short Hs[2][128 * LDSS];
    __shared__ __align__(16) unsigned short Ws[2][64 * LDSS];
    const int tid  = threadIdx.x;
    const int lane = tid & 31;
    const int wid  = tid >> 5;
    const int wm   = wid & 3;
    const int wn   = wid >> 2;

    v8f acc[2][2];
#pragma unroll
    for (int mi = 0; mi < 2; ++mi)
#pragma unroll
        for (int ni = 0; ni < 2; ++ni) acc[mi][ni] = zero8();

    uint4 hv[2];
    float wv[2][4];
    auto loadG = [&](int k) {
#pragma unroll
        for (int l = 0; l < 2; ++l) {   // hidden tile: bf16 already, uint4 = 8 elems
            const int idx = tid + l * 256;
            const int row = idx >> 2;
            const int c8  = (idx & 3) * 8;
            hv[l] = *reinterpret_cast<const uint4*>(
                hidden + ((size_t)e * CAP + m0 + row) * HP + k + c8);
        }
#pragma unroll
        for (int l = 0; l < 2; ++l) {   // w_down tile: rows = d dims; K-guard vs HH
            const int idx = tid + l * 256;
            const int row = idx >> 3;
            const int c4  = (idx & 7) * 4;
            const float* wr = w_down + ((size_t)e * DM + n0 + row) * HH;
#pragma unroll
            for (int j = 0; j < 4; ++j) {
                const int kk = k + c4 + j;
                wv[l][j] = (kk < HH) ? wr[kk] : 0.0f;
            }
        }
    };
    auto storeL = [&](int buf) {
#pragma unroll
        for (int l = 0; l < 2; ++l) {
            const int idx = tid + l * 256;
            *reinterpret_cast<uint4*>(&Hs[buf][(idx >> 2) * LDSS + (idx & 3) * 8]) = hv[l];
        }
#pragma unroll
        for (int l = 0; l < 2; ++l) {
            const int idx = tid + l * 256;
            store_bf16x4(&Ws[buf][(idx >> 3) * LDSS + (idx & 7) * 4],
                         wv[l][0], wv[l][1], wv[l][2], wv[l][3]);
        }
    };

    loadG(0);
    storeL(0);
    int buf = 0;
    constexpr int NT = HP / 32;
    for (int kt = 0; kt < NT; ++kt) {
        __syncthreads();
        const bool more = (kt + 1 < NT);
        if (more) loadG((kt + 1) * 32);
        const v16bf a0 = load_frag(Hs[buf], wm * 32,      lane);
        const v16bf a1 = load_frag(Hs[buf], wm * 32 + 16, lane);
        const v16bf b0 = load_frag(Ws[buf], wn * 32,      lane);
        const v16bf b1 = load_frag(Ws[buf], wn * 32 + 16, lane);
        acc[0][0] = wmma_bf16(a0, b0, acc[0][0]);
        acc[0][1] = wmma_bf16(a0, b1, acc[0][1]);
        acc[1][0] = wmma_bf16(a1, b0, acc[1][0]);
        acc[1][1] = wmma_bf16(a1, b1, acc[1][1]);
        if (more) storeL(buf ^ 1);
        buf ^= 1;
    }

#pragma unroll
    for (int mi = 0; mi < 2; ++mi) {
        const int rbase = m0 + wm * 32 + mi * 16 + ((lane >> 4) << 3);
#pragma unroll
        for (int ni = 0; ni < 2; ++ni) {
            const int cb = n0 + wn * 32 + ni * 16 + (lane & 15);
#pragma unroll
            for (int i = 0; i < 8; ++i)
                down_out[((size_t)e * CAP + rbase + i) * DM + cb] = acc[mi][ni][i];
        }
    }
}

// ---------------- 6. combine: out[t] = w0*row(e0,pos0) + w1*row(e1,pos1) --------
__global__ __launch_bounds__(256) void combine_kernel(
        const float* __restrict__ down_out, const int* __restrict__ slot_e,
        const int* __restrict__ slot_pos, const float* __restrict__ slot_w,
        float* __restrict__ out) {
    const size_t idx = (size_t)blockIdx.x * 256 + threadIdx.x;   // over T*D/4
    const int t = (int)(idx >> 8);            // D/4 == 256 float4 per token
    const int c = (int)(idx & 255) << 2;
    const int   e0 = slot_e[2 * t],     e1 = slot_e[2 * t + 1];
    const int   p0 = slot_pos[2 * t],   p1 = slot_pos[2 * t + 1];
    const float w0 = slot_w[2 * t],     w1 = slot_w[2 * t + 1];
    const float4 a = *reinterpret_cast<const float4*>(down_out + (((size_t)e0 * CAP + p0) << 10) + c);
    const float4 b = *reinterpret_cast<const float4*>(down_out + (((size_t)e1 * CAP + p1) << 10) + c);
    float4 o;
    o.x = w0 * a.x + w1 * b.x;
    o.y = w0 * a.y + w1 * b.y;
    o.z = w0 * a.z + w1 * b.z;
    o.w = w0 * a.w + w1 * b.w;
    *reinterpret_cast<float4*>(out + ((size_t)t << 10) + c) = o;
}

// ---------------- launch ----------------
extern "C" void kernel_launch(void* const* d_in, const int* in_sizes, int n_in,
                              void* d_out, int out_size, void* d_ws, size_t ws_size,
                              hipStream_t stream) {
    const float* x      = (const float*)d_in[0];
    const float* gate_w = (const float*)d_in[1];
    const float* w_gate = (const float*)d_in[2];
    const float* w_up   = (const float*)d_in[3];
    const float* w_down = (const float*)d_in[4];
    float* out = (float*)d_out;

    char* ws = (char*)d_ws;
    constexpr size_t HIDDEN_B  = (size_t)NE * CAP * HP * 2;        // 45,088,768
    constexpr size_t DOWN_B    = (size_t)NE * CAP * DM * 4;        // 33,554,432
    unsigned short* hidden   = (unsigned short*)ws;
    float*          down_out = (float*)(ws + HIDDEN_B);
    char* meta = ws + HIDDEN_B + DOWN_B;
    int*   topk_idx = (int*)  (meta);
    float* topk_p   = (float*)(meta + 1 * 65536);
    int*   keep     = (int*)  (meta + 2 * 65536);
    int*   slot_e   = (int*)  (meta + 3 * 65536);
    int*   slot_pos = (int*)  (meta + 4 * 65536);
    float* slot_w   = (float*)(meta + 5 * 65536);
    int*   tok_list = (int*)  (meta + 6 * 65536);

    router_topk_kernel  <<<TOK / 8, 256, 0, stream>>>(x, gate_w, topk_idx, topk_p);
    capacity_keep_kernel<<<TK / 256, 256, 0, stream>>>(topk_idx, topk_p, keep, slot_w);
    init_lists_kernel   <<<(NE * CAP) / 256, 256, 0, stream>>>(tok_list);
    positions_kernel    <<<TK / 256, 256, 0, stream>>>(topk_idx, keep, slot_e, slot_pos, tok_list);
    moe_gate_up_kernel  <<<dim3(HP / 64, CAP / 128, NE), 256, 0, stream>>>(x, w_gate, w_up, tok_list, hidden);
    moe_down_kernel     <<<dim3(DM / 64, CAP / 128, NE), 256, 0, stream>>>(hidden, w_down, down_out);
    combine_kernel      <<<(TOK * DM / 4) / 256, 256, 0, stream>>>(down_out, slot_e, slot_pos, slot_w, out);
}
